// mLSTMRecurrentKernelCell_45475113730286
// MI455X (gfx1250) — matrix operand order repaired and checked by reading, not compiled, not run on hardware
//
#include <hip/hip_runtime.h>
#include <hip/hip_bf16.h>

// ---------------------------------------------------------------------------
// Chunkwise-parallel mLSTM for MI455X (gfx1250, wave32), L=32 chunks.
//  - bf16 WMMA 16x16x32 (f32 accumulate); every chunk-K GEMM is exactly K=32
//  - C state in f32 VGPR accumulators per wave, bf16 shadow in LDS (N-major)
//  - all WMMA operand loads are lane-contiguous; rows padded for LDS banking
//  - exact reference gate stabilization (running max m) in f32
// ---------------------------------------------------------------------------

#define B_SZ 2
#define NH_SZ 8
#define S_SZ 1024
#define DQK_SZ 256
#define DV_SZ 512
#define L_CH 32                 // chunk length (two WMMA M tiles, K=32 native)
#define W_V 64                  // DV columns per workgroup
#define NDV (DV_SZ / W_V)       // 8
#define N_CHUNK (S_SZ / L_CH)   // 32
#define NTHR 256                // 8 waves
#define EPS_V 1e-6f
#define INV_SQRT_D 0.0625f      // 1/sqrt(256)

// padded leading dimensions (halves) to spread lane starts across LDS banks
#define QS_LD  (DQK_SZ + 8)     // 264: 528B row stride -> lane starts 4 banks apart
#define KS_LD  (DQK_SZ + 8)     // 264
#define CBF_LD (DQK_SZ + 16)    // 272: 544B row stride
#define KST_LD (L_CH + 8)       // 40:  80B row stride
#define VST_LD (L_CH + 8)       // 40
#define SD_LD  (L_CH + 8)       // 40

typedef __bf16 bf16;
typedef __attribute__((ext_vector_type(16))) __bf16 v16bf;
typedef __attribute__((ext_vector_type(8)))  float  v8f;

struct SMem {
  bf16 CbfT[W_V * CBF_LD];    // bf16 shadow of C, N-major: CbfT[n][row]   ~34KB
  bf16 qs[L_CH * QS_LD];      // (Q/sqrt(d)) chunk [t][d]                  ~17KB
  bf16 ks[L_CH * KS_LD];      // K chunk [t][d] (B-operand of S GEMM)      ~17KB
  bf16 ksT[DQK_SZ * KST_LD];  // K chunk transposed [d][t]                 ~20KB
  bf16 vsT[W_V * VST_LD];     // V chunk slice transposed [n][t]           ~5KB
  bf16 sd[L_CH * SD_LD];      // masked decayed scores (S .* D) [t][s]     ~2.5KB
  float nvec[DQK_SZ];         // running normalizer state n
  float iraw[L_CH], fraw[L_CH];
  float ia[L_CH];             // i_s - F_s
  float Ar[L_CH];             // running max A_t
  float cint[L_CH];           // exp(m_old - A_t)
  float enm[L_CH];            // exp(-m_t)
  float g[L_CH];              // exp(ia_s - A_L) end-of-chunk input gates
  float qn_intra[L_CH];       // rowsum(S .* D)
  float dotraw[L_CH];         // qs_t . n_old
  float ms;                   // running max m
  float cC;                   // exp(m_old - A_L) chunk decay on C, n
};

__device__ __forceinline__ v8f wmma_bf16(v16bf a, v16bf b, v8f c) {
  return __builtin_amdgcn_wmma_f32_16x16x32_bf16(false, a, false, b,
                                                 (short)0, c, false, false);
}

// A fragment: 16xK row-major bf16 source, k-block kb covers K=[kb*32,kb*32+32)
// ISA layout: lanes0-15 M=0..15 (K +0), lanes16-31 M=0..15 (K +8);
//             elems 0..7 -> K base 0, elems 8..15 -> K base 16.
__device__ __forceinline__ v16bf load_a_rm(const bf16* A, int ld, int kb, int lane) {
  const int m = lane & 15, hi = (lane >> 4) & 1;
  const bf16* p = A + m * ld + kb * 32 + hi * 8;
  v16bf a;
#pragma unroll
  for (int j = 0; j < 8; ++j) a[j] = p[j];
#pragma unroll
  for (int j = 0; j < 8; ++j) a[8 + j] = p[16 + j];
  return a;
}

// B fragment from N-major source (row n holds that column's K run, contiguous).
__device__ __forceinline__ v16bf load_b_nxk(const bf16* Bt, int ld, int kb, int lane) {
  const int n = lane & 15, hi = (lane >> 4) & 1;
  const bf16* p = Bt + n * ld + kb * 32 + hi * 8;
  v16bf b;
#pragma unroll
  for (int j = 0; j < 8; ++j) b[j] = p[j];
#pragma unroll
  for (int j = 0; j < 8; ++j) b[8 + j] = p[16 + j];
  return b;
}

// On-the-fly A fragment for the C update: A[m][s] = k_s[m] * g[s], full K=32.
// Reads ksT (d-major) so each half is an 8-half contiguous run.
__device__ __forceinline__ v16bf load_a_kg32(const bf16* ksT, const float* g, int mi, int lane) {
  const int m = mi * 16 + (lane & 15), hi = (lane >> 4) & 1;
  const bf16* p = ksT + m * KST_LD;
  v16bf a;
#pragma unroll
  for (int j = 0; j < 8; ++j) {
    const int s = hi * 8 + j;
    a[j] = (bf16)((float)p[s] * g[s]);
  }
#pragma unroll
  for (int j = 0; j < 8; ++j) {
    const int s = 16 + hi * 8 + j;
    a[8 + j] = (bf16)((float)p[s] * g[s]);
  }
  return a;
}

__global__ __launch_bounds__(NTHR)
void mlstm_chunkwise_wmma(const float* __restrict__ qp, const float* __restrict__ kp,
                          const float* __restrict__ vp, const float* __restrict__ ipre,
                          const float* __restrict__ fpre, float* __restrict__ out) {
  __shared__ SMem sm;
  const int tid = threadIdx.x;
  const int lane = tid & 31;
  const int wid = tid >> 5;
  const int bh = blockIdx.x / NDV;       // fused (b, nh) index
  const int dvblk = blockIdx.x % NDV;
  const int dv0 = dvblk * W_V;

  const size_t qkbase = (size_t)bh * S_SZ * DQK_SZ;
  const size_t vbase  = (size_t)bh * S_SZ * DV_SZ + dv0;
  const size_t ifbase = (size_t)bh * S_SZ;

  float* hout = out;                                          // [B,NH,S,DV]
  float* Cout = out + (size_t)B_SZ * NH_SZ * S_SZ * DV_SZ;    // [B,NH,DQK,DV]
  float* nout = Cout + (size_t)B_SZ * NH_SZ * DQK_SZ * DV_SZ; // [B,NH,DQK]
  float* mout = nout + (size_t)B_SZ * NH_SZ * DQK_SZ;         // [B,NH]

  // Wave-private f32 C state: wave wid owns M-tiles [wid*2, wid*2+2) x 4 N-tiles.
  // Cacc[tl]: tl = mi2*4 + w2  (mi = wid*2 + mi2, cols [w2*16, w2*16+16))
  v8f Cacc[8];
#pragma unroll
  for (int tl = 0; tl < 8; ++tl)
#pragma unroll
    for (int r = 0; r < 8; ++r) Cacc[tl][r] = 0.0f;

  for (int idx = tid; idx < W_V * CBF_LD; idx += NTHR) sm.CbfT[idx] = (bf16)0.0f;
  // upper-right score tile (rows 0..15, cols 16..31) is always fully masked:
  // zero it once, it is never written afterwards.
  if (tid < 256) sm.sd[(tid >> 4) * SD_LD + 16 + (tid & 15)] = (bf16)0.0f;
  if (tid < DQK_SZ) sm.nvec[tid] = 0.0f;
  if (tid == 0) { sm.ms = 0.0f; sm.cC = 1.0f; }
  __syncthreads();

  for (int ch = 0; ch < N_CHUNK; ++ch) {
    const int c0 = ch * L_CH;

    // ---- Phase A: stage chunk into LDS (coalesced), prefetch next chunk ----
    for (int idx = tid; idx < L_CH * DQK_SZ; idx += NTHR) {
      const int t = idx >> 8, d = idx & (DQK_SZ - 1);
      const float qv = qp[qkbase + (size_t)(c0 + t) * DQK_SZ + d];
      const float kv = kp[qkbase + (size_t)(c0 + t) * DQK_SZ + d];
      sm.qs[t * QS_LD + d] = (bf16)(qv * INV_SQRT_D);
      const bf16 kb16 = (bf16)kv;
      sm.ks[t * KS_LD + d] = kb16;
      sm.ksT[d * KST_LD + t] = kb16;
    }
    for (int idx = tid; idx < L_CH * W_V; idx += NTHR) {
      const int t = idx >> 6, j = idx & (W_V - 1);
      sm.vsT[j * VST_LD + t] = (bf16)(vp[vbase + (size_t)(c0 + t) * DV_SZ + j]);
    }
    if (tid < L_CH) {
      sm.iraw[tid] = ipre[ifbase + c0 + tid];
      sm.fraw[tid] = fpre[ifbase + c0 + tid];
      sm.dotraw[tid] = 0.0f;
      sm.qn_intra[tid] = 0.0f;
    }
    if (ch + 1 < N_CHUNK) {
      __builtin_prefetch(qp + qkbase + (size_t)(c0 + L_CH) * DQK_SZ + tid * 32, 0, 0);
      __builtin_prefetch(kp + qkbase + (size_t)(c0 + L_CH) * DQK_SZ + tid * 32, 0, 0);
      __builtin_prefetch(vp + vbase + (size_t)(c0 + L_CH + (tid >> 3)) * DV_SZ + (tid & 7) * 8, 0, 0);
    }
    __syncthreads();

    // ---- Phase B: gate scan (thread0) | S tiles (waves0-2) | H_inter (waves4-7)
    v8f accS;
    v8f accH[2];
#pragma unroll
    for (int i = 0; i < 8; ++i) { accS[i] = 0.0f; accH[0][i] = 0.0f; accH[1][i] = 0.0f; }

    if (tid == 0) {
      const float mo = sm.ms;
      float F = 0.0f, A = mo;
#pragma unroll 1
      for (int s = 0; s < L_CH; ++s) {
        const float fv = sm.fraw[s];
        const float fl = fminf(fv, 0.0f) - log1pf(expf(-fabsf(fv))); // log_sigmoid
        F += fl;
        const float iav = sm.iraw[s] - F;
        A = fmaxf(A, iav);
        sm.ia[s] = iav;
        sm.Ar[s] = A;
        sm.cint[s] = expf(mo - A);       // <= 1
        sm.enm[s] = expf(-(F + A));      // exp(-m_t)
      }
      sm.cC = expf(mo - A);
#pragma unroll 1
      for (int s = 0; s < L_CH; ++s) sm.g[s] = expf(sm.ia[s] - A); // <= 1
      sm.ms = F + A;                     // m after this chunk
    }
    if (wid < 3) {
      // tiles (ti,tj): wave0 -> (0,0), wave1 -> (1,0), wave2 -> (1,1)
      const int ti = (wid + 1) >> 1, tj = wid >> 1;
      const bf16* aq = sm.qs + ti * 16 * QS_LD;
      const bf16* bk = sm.ks + tj * 16 * KS_LD;
#pragma unroll
      for (int kb = 0; kb < DQK_SZ / 32; ++kb)
        accS = wmma_bf16(load_a_rm(aq, QS_LD, kb, lane),
                         load_b_nxk(bk, KS_LD, kb, lane), accS);
    } else if (wid >= 4) {
      const int w2 = wid - 4;
      const bf16* bc = sm.CbfT + (w2 * 16) * CBF_LD;
#pragma unroll
      for (int kb = 0; kb < DQK_SZ / 32; ++kb) {
        const v16bf vb = load_b_nxk(bc, CBF_LD, kb, lane);  // shared by both M tiles
        accH[0] = wmma_bf16(load_a_rm(sm.qs, QS_LD, kb, lane), vb, accH[0]);
        accH[1] = wmma_bf16(load_a_rm(sm.qs + 16 * QS_LD, QS_LD, kb, lane), vb, accH[1]);
      }
    }
    __syncthreads();

    // ---- Phase C: mask/decay S | scale H_inter | C state update | qn dots ----
    if (wid < 3) {
      const int ti = (wid + 1) >> 1, tj = wid >> 1;
      const int sloc = tj * 16 + (lane & 15);
      const int hiw = (lane >> 4) & 1;
#pragma unroll
      for (int r = 0; r < 8; ++r) {
        const int t = ti * 16 + r + hiw * 8;
        float val = (sloc <= t) ? accS[r] * expf(sm.ia[sloc] - sm.Ar[t]) : 0.0f;
        sm.sd[t * SD_LD + sloc] = (bf16)val;
        float rs = val;
#pragma unroll
        for (int off = 1; off < 16; off <<= 1) rs += __shfl_xor(rs, off, 16);
        if ((lane & 15) == 0) atomicAdd(&sm.qn_intra[t], rs);
      }
    }
    if (wid >= 4) {
      const int hiw = (lane >> 4) & 1;
#pragma unroll
      for (int ti = 0; ti < 2; ++ti)
#pragma unroll
        for (int r = 0; r < 8; ++r) accH[ti][r] *= sm.cint[ti * 16 + r + hiw * 8];
    }
    {
      // C = cC * C + (K^T diag(g)) V on this wave's 8 owned tiles (K=32 exact).
      const float cCv = sm.cC;
      const int hiw = (lane >> 4) & 1;
      const int nN = lane & 15;
      const v16bf va0 = load_a_kg32(sm.ksT, sm.g, wid * 2 + 0, lane);
      const v16bf va1 = load_a_kg32(sm.ksT, sm.g, wid * 2 + 1, lane);
      const int row0base = (wid * 2 + 0) * 16 + hiw * 8;
      const int row1base = (wid * 2 + 1) * 16 + hiw * 8;
#pragma unroll
      for (int w2 = 0; w2 < 4; ++w2) {
        const v16bf vb = load_b_nxk(sm.vsT + (w2 * 16) * VST_LD, VST_LD, 0, lane);
#pragma unroll
        for (int r = 0; r < 8; ++r) Cacc[w2][r] *= cCv;
        Cacc[w2] = wmma_bf16(va0, vb, Cacc[w2]);
#pragma unroll
        for (int r = 0; r < 8; ++r) Cacc[4 + w2][r] *= cCv;
        Cacc[4 + w2] = wmma_bf16(va1, vb, Cacc[4 + w2]);
        // N-major shadow write-back: 8 contiguous halves per lane (vectorizes)
        bf16* dst = sm.CbfT + (w2 * 16 + nN) * CBF_LD;
#pragma unroll
        for (int r = 0; r < 8; ++r) dst[row0base + r] = (bf16)Cacc[w2][r];
#pragma unroll
        for (int r = 0; r < 8; ++r) dst[row1base + r] = (bf16)Cacc[4 + w2][r];
      }
    }
    {
      // denominator inter-term partials: dotraw[t] += qs_t . n_old
      const int t = tid >> 3, l8 = tid & 7;
      float part = 0.0f;
#pragma unroll
      for (int j = 0; j < 32; ++j) {
        const int d = l8 * 32 + j;
        part += (float)sm.qs[t * QS_LD + d] * sm.nvec[d];
      }
      atomicAdd(&sm.dotraw[t], part);
    }
    __syncthreads();

    // ---- Phase D: H += (S.*D)V, write h (waves4-7) | n update (waves0-3) ----
    if (wid >= 4) {
      const int w2 = wid - 4;
      const v16bf vb = load_b_nxk(sm.vsT + (w2 * 16) * VST_LD, VST_LD, 0, lane);
      accH[0] = wmma_bf16(load_a_rm(sm.sd, SD_LD, 0, lane), vb, accH[0]);
      accH[1] = wmma_bf16(load_a_rm(sm.sd + 16 * SD_LD, SD_LD, 0, lane), vb, accH[1]);
      const int nN = lane & 15;
      const int hiw = (lane >> 4) & 1;
#pragma unroll
      for (int ti = 0; ti < 2; ++ti)
#pragma unroll
        for (int r = 0; r < 8; ++r) {
          const int t = ti * 16 + r + hiw * 8;
          const float qn = sm.qn_intra[t] + sm.cint[t] * sm.dotraw[t];
          const float den = fmaxf(fabsf(qn), sm.enm[t]) + EPS_V;
          const float rden = __builtin_amdgcn_rcpf(den);  // ~1ulp, plenty for a stabilizer
          hout[(size_t)bh * S_SZ * DV_SZ + (size_t)(c0 + t) * DV_SZ + dv0 + w2 * 16 + nN] =
              accH[ti][r] * rden;
        }
    } else {
      for (int d = tid; d < DQK_SZ; d += 128) {
        const bf16* p = sm.ksT + d * KST_LD;
        float nv = sm.cC * sm.nvec[d];
#pragma unroll
        for (int s = 0; s < L_CH; ++s) nv += sm.g[s] * (float)p[s];
        sm.nvec[d] = nv;
      }
    }
    __syncthreads();
  }

  // ---- Final state write-out ----
  {
    const int hiw = (lane >> 4) & 1;
    const int nN = lane & 15;
#pragma unroll
    for (int tl = 0; tl < 8; ++tl) {
      const int mi = wid * 2 + (tl >> 2);
      const int w2 = tl & 3;
#pragma unroll
      for (int r = 0; r < 8; ++r) {
        const int row = mi * 16 + r + hiw * 8;
        Cout[((size_t)bh * DQK_SZ + row) * DV_SZ + dv0 + w2 * 16 + nN] = Cacc[tl][r];
      }
    }
  }
  if (dvblk == 0) {
    if (tid < DQK_SZ) nout[(size_t)bh * DQK_SZ + tid] = sm.nvec[tid];
    if (tid == 0) mout[bh] = sm.ms;
  }
}

extern "C" void kernel_launch(void* const* d_in, const int* in_sizes, int n_in,
                              void* d_out, int out_size, void* d_ws, size_t ws_size,
                              hipStream_t stream) {
  (void)in_sizes; (void)n_in; (void)out_size; (void)d_ws; (void)ws_size;
  const float* q = (const float*)d_in[0];
  const float* k = (const float*)d_in[1];
  const float* v = (const float*)d_in[2];
  const float* ip = (const float*)d_in[3];
  const float* fp = (const float*)d_in[4];
  float* out = (float*)d_out;

  dim3 grid(B_SZ * NH_SZ * NDV);  // 128 workgroups
  dim3 block(NTHR);               // 8 wave32 waves
  mlstm_chunkwise_wmma<<<grid, block, 0, stream>>>(q, k, v, ip, fp, out);
}
